// MinimalRNNSimple_33595234189741
// MI455X (gfx1250) — matrix operand order repaired and checked by reading, not compiled, not run on hardware
//
#include <hip/hip_runtime.h>
#include <math.h>
#include <stddef.h>

// Problem dims (fixed by the reference)
#define B_  16
#define T_  2048
#define E_  1024
#define S_  1024
#define M_  (B_ * T_)      // 32768 rows for all three big GEMMs
#define EPS 1e-5f

// ---------------------------------------------------------------- types ----
typedef __attribute__((ext_vector_type(16))) __bf16          v16bf;
typedef __attribute__((ext_vector_type(8)))  float           v8f;
typedef __attribute__((ext_vector_type(8)))  unsigned short  u16x8;
typedef __attribute__((vector_size(16)))     int             v4i;

union FragU {
    u16x8 h[2];
    v16bf v;
};

// ------------------------------------------------------------- helpers ----
__device__ __host__ inline unsigned short f2bf(float f) {
    unsigned u = __builtin_bit_cast(unsigned, f);
    unsigned r = u + 0x7FFFu + ((u >> 16) & 1u);   // RNE truncate to bf16
    return (unsigned short)(r >> 16);
}
__device__ __host__ inline float bf2f(unsigned short h) {
    unsigned u = ((unsigned)h) << 16;
    return __builtin_bit_cast(float, u);
}

// 3-term bf16 split product: (ahi+alo)*(bhi+blo) ~ ahi*bhi + ahi*blo + alo*bhi
__device__ inline v8f wmma3(const FragU& ah, const FragU& al,
                            const FragU& bh, const FragU& bl, v8f acc) {
#if defined(__HIP_DEVICE_COMPILE__)
    acc = __builtin_amdgcn_wmma_f32_16x16x32_bf16(false, ah.v, false, bh.v,
                                                  (short)0, acc, false, false);
    acc = __builtin_amdgcn_wmma_f32_16x16x32_bf16(false, ah.v, false, bl.v,
                                                  (short)0, acc, false, false);
    acc = __builtin_amdgcn_wmma_f32_16x16x32_bf16(false, al.v, false, bh.v,
                                                  (short)0, acc, false, false);
#else
    (void)ah; (void)al; (void)bh; (void)bl;
#endif
    return acc;
}

// 16-byte global -> LDS copy: async DMA (ASYNCcnt) if the toolchain exposes
// the gfx1250 builtin, else VGPR round-trip fallback.
#if defined(__HIP_DEVICE_COMPILE__) && \
    __has_builtin(__builtin_amdgcn_global_load_async_to_lds_b128)
#define HAS_ASYNC_LDS 1
#else
#define HAS_ASYNC_LDS 0
#endif

__device__ inline void cp16(unsigned short* dst, const unsigned short* src) {
#if HAS_ASYNC_LDS
    __builtin_amdgcn_global_load_async_to_lds_b128(
        (__attribute__((address_space(1))) v4i*)(unsigned short*)src,
        (__attribute__((address_space(3))) v4i*)dst, 0, 0);
#else
    *(u16x8*)dst = *(const u16x8*)src;
#endif
}

__device__ inline void cp16_wait() {
#if HAS_ASYNC_LDS
#if __has_builtin(__builtin_amdgcn_s_wait_asynccnt)
    __builtin_amdgcn_s_wait_asynccnt(0);
#else
    asm volatile("s_wait_asynccnt 0" ::: "memory");
#endif
#endif
}

// ------------------------------------------------------- split kernels ----
__global__ void split_kernel(const float* __restrict__ in,
                             unsigned short* __restrict__ hi,
                             unsigned short* __restrict__ lo, size_t n) {
    size_t i = (size_t)blockIdx.x * blockDim.x + threadIdx.x;
    size_t stride = (size_t)gridDim.x * blockDim.x;
    for (; i < n; i += stride) {
        float f = in[i];
        unsigned short h = f2bf(f);
        hi[i] = h;
        lo[i] = f2bf(f - bf2f(h));
    }
}

// in is R x C row-major; out is transposed (C x R) hi/lo
__global__ void split_transpose_kernel(const float* __restrict__ in,
                                       unsigned short* __restrict__ hiT,
                                       unsigned short* __restrict__ loT,
                                       int R, int C) {
    int idx = blockIdx.x * blockDim.x + threadIdx.x;
    if (idx >= R * C) return;
    int r = idx / C, c = idx % C;
    float f = in[idx];
    unsigned short h = f2bf(f);
    hiT[(size_t)c * R + r] = h;
    loT[(size_t)c * R + r] = f2bf(f - bf2f(h));
}

__global__ void init_h_kernel(float* __restrict__ h,
                              unsigned short* __restrict__ hhi,
                              unsigned short* __restrict__ hlo,
                              unsigned* __restrict__ cnt) {
    int i = blockIdx.x * blockDim.x + threadIdx.x;
    if (i < B_ * S_) { h[i] = 0.f; hhi[i] = 0; hlo[i] = 0; }
    if (i == 0) *cnt = 0u;
}

// --------------------------------------------------- big GEMM (bf16x3) ----
// C[M,N] = epilogue( sum_k A[M,K]*B[N,K]^T + bias[N] )
// A,B given as bf16 hi/lo pairs, K-contiguous.
// EPI==0: tanh -> write hi/lo bf16 split; EPI==1: write fp32.
// Tiling: CTA tile 64(M) x 128(N), BK=32. 8 waves = 4(M) x 2(N),
// each wave computes a 16x64 strip (4 accumulators).
// LDS tiles are double-buffered and filled by async global->LDS DMA when
// available: one barrier + one ASYNCcnt wait per K-step, next tile's copies
// overlap the current tile's WMMAs.
#define LDS_STRIDE 48   // ushorts per row (32 data + 16 pad, keeps 16B align)

template <int EPI>
__global__ __launch_bounds__(256)
void gemm_bf16x3(const unsigned short* __restrict__ Ahi,
                 const unsigned short* __restrict__ Alo,
                 const unsigned short* __restrict__ Bhi,
                 const unsigned short* __restrict__ Blo,
                 const float* __restrict__ bias,
                 float* __restrict__ Cf,
                 unsigned short* __restrict__ Chi,
                 unsigned short* __restrict__ Clo,
                 int Ndim, int Kdim) {
    __shared__ unsigned short sAhi[2][64 * LDS_STRIDE],  sAlo[2][64 * LDS_STRIDE];
    __shared__ unsigned short sBhi[2][128 * LDS_STRIDE], sBlo[2][128 * LDS_STRIDE];

    const int tid  = threadIdx.x;
    const int lane = tid & 31;
    const int wave = tid >> 5;
    const int wm   = wave & 3;    // M subtile 0..3
    const int wn   = wave >> 2;   // N group 0..1
    const int nBlocks = Ndim / 128;
    const int nb = blockIdx.x % nBlocks;
    const int mb = blockIdx.x / nBlocks;
    const int rowBase = mb * 64;
    const int colBase = nb * 128;

    const int l15  = lane & 15;
    const int h16  = lane >> 4;

    // per-thread staging chunk coordinates (fixed for the whole kernel)
    const int ar = tid >> 2, ac = tid & 3;            // A: 256 chunks of 8

    v8f acc[4];
    #pragma unroll
    for (int s = 0; s < 4; ++s)
        acc[s] = (v8f){0.f,0.f,0.f,0.f,0.f,0.f,0.f,0.f};

    // issue one K-step tile (A 64x32 + B 128x32, hi & lo) into buffer `buf`
    auto issue_stage = [&](int k, int buf) {
        size_t ga = (size_t)(rowBase + ar) * Kdim + k + ac * 8;
        cp16(&sAhi[buf][ar * LDS_STRIDE + ac * 8], Ahi + ga);
        cp16(&sAlo[buf][ar * LDS_STRIDE + ac * 8], Alo + ga);
        #pragma unroll
        for (int it = 0; it < 2; ++it) {
            int idx = it * 256 + tid;
            int br = idx >> 2, bc = idx & 3;
            size_t gb = (size_t)(colBase + br) * Kdim + k + bc * 8;
            cp16(&sBhi[buf][br * LDS_STRIDE + bc * 8], Bhi + gb);
            cp16(&sBlo[buf][br * LDS_STRIDE + bc * 8], Blo + gb);
        }
    };

    issue_stage(0, 0);
    int cur = 0;
    for (int k = 0; k < Kdim; k += 32) {
        cp16_wait();          // this wave's copies into buf[cur] are done
        __syncthreads();      // everyone's copies done; prev reads retired
        if (k + 32 < Kdim) issue_stage(k + 32, cur ^ 1);   // overlap next tile

        // ---- A fragment (16x32): lane<16 -> K[0..7]+K[16..23], lane>=16 -> +8
        FragU ahi, alo;
        {
            int m = wm * 16 + l15;
            int koff = h16 * 8;
            ahi.h[0] = *(const u16x8*)&sAhi[cur][m * LDS_STRIDE + koff];
            ahi.h[1] = *(const u16x8*)&sAhi[cur][m * LDS_STRIDE + 16 + koff];
            alo.h[0] = *(const u16x8*)&sAlo[cur][m * LDS_STRIDE + koff];
            alo.h[1] = *(const u16x8*)&sAlo[cur][m * LDS_STRIDE + 16 + koff];
        }
        // ---- 4 N-subtiles, 3 WMMAs each
        #pragma unroll
        for (int s = 0; s < 4; ++s) {
            FragU bhi, blo;
            int n  = wn * 64 + s * 16 + l15;
            int kb = h16 * 8;
            bhi.h[0] = *(const u16x8*)&sBhi[cur][n * LDS_STRIDE + kb];
            bhi.h[1] = *(const u16x8*)&sBhi[cur][n * LDS_STRIDE + kb + 16];
            blo.h[0] = *(const u16x8*)&sBlo[cur][n * LDS_STRIDE + kb];
            blo.h[1] = *(const u16x8*)&sBlo[cur][n * LDS_STRIDE + kb + 16];
            acc[s] = wmma3(ahi, alo, bhi, blo, acc[s]);
        }
        cur ^= 1;
    }

    // ---- epilogue: C layout — lane holds col=l15, rows v + 8*(lane>=16)
    #pragma unroll
    for (int s = 0; s < 4; ++s) {
        int col  = colBase + wn * 64 + s * 16 + l15;
        float bc = bias[col];
        #pragma unroll
        for (int v = 0; v < 8; ++v) {
            int m = rowBase + wm * 16 + v + h16 * 8;
            float val = acc[s][v] + bc;
            size_t o = (size_t)m * Ndim + col;
            if (EPI == 0) {
                float tv = tanhf(val);
                unsigned short hh = f2bf(tv);
                Chi[o] = hh;
                Clo[o] = f2bf(tv - bf2f(hh));
            } else {
                Cf[o] = val;
            }
        }
    }
}

// ---------------------------------------------- persistent recurrence -----
// grid = NWG(16) blocks x 128 threads (4 waves); each wave owns 16 columns.
// Per-WG slice of U_h^T (64 cols x 1024 K, hi+lo bf16) is cached in LDS
// (258 KB of the WGP's 320 KB) once, then each step does:
//   u = sigmoid(h @ U_h + zu_t); h = LN(u*h + (1-u)*z_t)*g_s + b_s
#define NWG 16
#define BSTRIDE 1032   // ushorts per cached column (1024 + 8 pad -> 4-bank skew)

__device__ inline void grid_barrier(unsigned* cnt, unsigned nb, unsigned phase) {
    __threadfence();
    __syncthreads();
    if (threadIdx.x == 0) {
        __hip_atomic_fetch_add(cnt, 1u, __ATOMIC_RELEASE, __HIP_MEMORY_SCOPE_AGENT);
        while (__hip_atomic_load(cnt, __ATOMIC_ACQUIRE, __HIP_MEMORY_SCOPE_AGENT)
               < nb * phase) {
#if defined(__HIP_DEVICE_COMPILE__)
            __builtin_amdgcn_s_sleep(1);
#endif
        }
    }
    __syncthreads();
}

__global__ __launch_bounds__(128)
void recurrence_kernel(const unsigned short* __restrict__ UhThi,
                       const unsigned short* __restrict__ UhTlo,
                       const unsigned short* __restrict__ zhi,
                       const unsigned short* __restrict__ zlo,
                       const float* __restrict__ zu,
                       const float* __restrict__ g_s,
                       const float* __restrict__ b_s,
                       float* __restrict__ h,
                       unsigned short* __restrict__ hhi,
                       unsigned short* __restrict__ hlo,
                       unsigned short* __restrict__ shi,
                       unsigned short* __restrict__ slo,
                       float* __restrict__ psum,   // [16][NWG][2]
                       unsigned* __restrict__ cnt) {
    extern __shared__ unsigned short smem[];
    unsigned short* sBh = smem;                    // [64][BSTRIDE]
    unsigned short* sBl = smem + 64 * BSTRIDE;

    const int tid  = threadIdx.x;
    const int lane = tid & 31;
    const int wave = tid >> 5;                     // 0..3
    const int wg   = blockIdx.x;                   // 0..15
    const int l15  = lane & 15;
    const int h16  = lane >> 4;
    const int colL = wave * 16 + l15;              // local column 0..63
    const int col  = wg * 64 + colL;               // global hidden column
    const float gs = g_s[col], bs = b_s[col];

    // ---- cache this WG's U_h^T slice (hi+lo) in LDS, fragment-ready layout
    for (int idx = tid; idx < 64 * 128; idx += 128) {  // 8-ushort chunks
        int cl = idx >> 7;        // local column 0..63
        int ck = idx & 127;       // chunk within K (128 * 8 = 1024)
        size_t g = (size_t)(wg * 64 + cl) * S_ + ck * 8;
        *(u16x8*)&sBh[cl * BSTRIDE + ck * 8] = *(const u16x8*)(UhThi + g);
        *(u16x8*)&sBl[cl * BSTRIDE + ck * 8] = *(const u16x8*)(UhTlo + g);
    }
    __syncthreads();

    __shared__ float ssum[4][16], ssq[4][16];
    __shared__ float smean[16], srs[16];
    unsigned phase = 1;

    for (int t = 0; t < T_; ++t) {
        // opaque 0: stops the compiler proving LDS/B loads t-invariant
        // (round 1 it hoisted all 32 k-steps of B frags and spilled to scratch)
        unsigned tkick = 0;
        asm volatile("" : "+v"(tkick));

        // ---------------- GEMM: acc(16 x this-wave's 16 cols) over K=1024
        v8f acc = (v8f){0.f,0.f,0.f,0.f,0.f,0.f,0.f,0.f};
        for (int k = 0; k < S_; k += 32) {
            FragU ahi, alo, bhi, blo;
            {   // A = h (16 x 1024), hi/lo bf16; L2-resident, rewritten each t
                int m = l15, koff = h16 * 8;
                const unsigned short* p = hhi + (size_t)m * S_ + k + koff;
                const unsigned short* q = hlo + (size_t)m * S_ + k + koff;
                ahi.h[0] = *(const u16x8*)p;
                ahi.h[1] = *(const u16x8*)(p + 16);
                alo.h[0] = *(const u16x8*)q;
                alo.h[1] = *(const u16x8*)(q + 16);
            }
            {   // B = U_h^T column slice from LDS
                unsigned o = (unsigned)colL * BSTRIDE + k + h16 * 8 + tkick;
                bhi.h[0] = *(const u16x8*)&sBh[o];
                bhi.h[1] = *(const u16x8*)&sBh[o + 16];
                blo.h[0] = *(const u16x8*)&sBl[o];
                blo.h[1] = *(const u16x8*)&sBl[o + 16];
            }
            acc = wmma3(ahi, alo, bhi, blo, acc);
        }

        // ---------------- elementwise gate + candidate, per-row partials
        float c[8], rsum[8], rsq[8];
        #pragma unroll
        for (int v = 0; v < 8; ++v) {
            int m = v + h16 * 8;                      // batch row
            size_t zidx = ((size_t)m * T_ + t) * S_ + col;
            float a  = acc[v] + zu[zidx];
            float u  = 1.f / (1.f + expf(-a));
            float zf = bf2f(zhi[zidx]) + bf2f(zlo[zidx]);
            float hp = h[(size_t)m * S_ + col];
            float cv = u * hp + (1.f - u) * zf;
            c[v] = cv;  rsum[v] = cv;  rsq[v] = cv * cv;
        }
        // butterfly over the 16 lanes of each half (sums this wave's 16 cols)
        #pragma unroll
        for (int mask = 1; mask < 16; mask <<= 1) {
            #pragma unroll
            for (int v = 0; v < 8; ++v) {
                rsum[v] += __shfl_xor(rsum[v], mask, 32);
                rsq[v]  += __shfl_xor(rsq[v],  mask, 32);
            }
        }
        if (l15 == 0) {
            #pragma unroll
            for (int v = 0; v < 8; ++v) {
                ssum[wave][v + h16 * 8] = rsum[v];
                ssq[wave][v + h16 * 8]  = rsq[v];
            }
        }
        __syncthreads();
        if (tid < 16) {   // deterministic per-WG partial (fixed order)
            float s1 = 0.f, s2 = 0.f;
            #pragma unroll
            for (int w = 0; w < 4; ++w) { s1 += ssum[w][tid]; s2 += ssq[w][tid]; }
            psum[(tid * NWG + wg) * 2 + 0] = s1;
            psum[(tid * NWG + wg) * 2 + 1] = s2;
        }
        grid_barrier(cnt, NWG, phase++);          // partials globally visible

        if (tid < 16) {   // deterministic global reduce (fixed order)
            float s1 = 0.f, s2 = 0.f;
            #pragma unroll
            for (int w = 0; w < NWG; ++w) {
                s1 += psum[(tid * NWG + w) * 2 + 0];
                s2 += psum[(tid * NWG + w) * 2 + 1];
            }
            float mean = s1 * (1.f / S_);
            float var  = s2 * (1.f / S_) - mean * mean;
            smean[tid] = mean;
            srs[tid]   = rsqrtf(var + EPS);
        }
        __syncthreads();

        // ---------------- layernorm + write h (fp32 + bf16 hi/lo) + states
        #pragma unroll
        for (int v = 0; v < 8; ++v) {
            int m = v + h16 * 8;
            float hn = (c[v] - smean[m]) * srs[m] * gs + bs;
            size_t hidx = (size_t)m * S_ + col;
            h[hidx] = hn;
            unsigned short hh = f2bf(hn);
            unsigned short hl = f2bf(hn - bf2f(hh));
            hhi[hidx] = hh;  hlo[hidx] = hl;
            size_t so = ((size_t)m * T_ + t) * S_ + col;
            shi[so] = hh;    slo[so] = hl;
        }
        grid_barrier(cnt, NWG, phase++);          // h visible for next step
    }
}

// -------------------------------------------------- final row layernorm ---
__global__ __launch_bounds__(256)
void layernorm_rows(float* __restrict__ x, const float* __restrict__ g,
                    const float* __restrict__ b) {
    __shared__ float s1[256], s2[256];
    const int row = blockIdx.x, tid = threadIdx.x;
    float vals[4], a = 0.f, a2 = 0.f;
    #pragma unroll
    for (int i = 0; i < 4; ++i) {
        float v = x[(size_t)row * E_ + tid + i * 256];
        vals[i] = v;  a += v;  a2 += v * v;
    }
    s1[tid] = a;  s2[tid] = a2;
    __syncthreads();
    for (int off = 128; off > 0; off >>= 1) {
        if (tid < off) { s1[tid] += s1[tid + off]; s2[tid] += s2[tid + off]; }
        __syncthreads();
    }
    float mean = s1[0] * (1.f / E_);
    float var  = s2[0] * (1.f / E_) - mean * mean;
    float rs   = rsqrtf(var + EPS);
    #pragma unroll
    for (int i = 0; i < 4; ++i) {
        int cidx = tid + i * 256;
        x[(size_t)row * E_ + cidx] = (vals[i] - mean) * rs * g[cidx] + b[cidx];
    }
}

// ----------------------------------------------------------- launcher -----
extern "C" void kernel_launch(void* const* d_in, const int* in_sizes, int n_in,
                              void* d_out, int out_size, void* d_ws, size_t ws_size,
                              hipStream_t stream) {
    (void)in_sizes; (void)n_in; (void)out_size; (void)ws_size;
    const float* x    = (const float*)d_in[0];
    const float* Wz   = (const float*)d_in[1];
    const float* bz   = (const float*)d_in[2];
    const float* U_h  = (const float*)d_in[3];
    const float* U_z  = (const float*)d_in[4];
    const float* b_u  = (const float*)d_in[5];
    const float* Wout = (const float*)d_in[6];
    const float* bout = (const float*)d_in[7];
    const float* g_s  = (const float*)d_in[8];
    const float* b_s  = (const float*)d_in[9];
    const float* g_o  = (const float*)d_in[10];
    const float* b_o  = (const float*)d_in[11];
    float* out = (float*)d_out;

    // ---- carve workspace
    char* w = (char*)d_ws;
    size_t off = 0;
    auto alloc = [&](size_t bytes) -> void* {
        void* p = w + off;
        off += (bytes + 255) & ~(size_t)255;
        return p;
    };
    unsigned short* Xhi  = (unsigned short*)alloc((size_t)M_ * E_ * 2);
    unsigned short* Xlo  = (unsigned short*)alloc((size_t)M_ * E_ * 2);
    unsigned short* Wzhi = (unsigned short*)alloc((size_t)S_ * E_ * 2);
    unsigned short* Wzlo = (unsigned short*)alloc((size_t)S_ * E_ * 2);
    unsigned short* UzThi = (unsigned short*)alloc((size_t)S_ * S_ * 2);
    unsigned short* UzTlo = (unsigned short*)alloc((size_t)S_ * S_ * 2);
    unsigned short* UhThi = (unsigned short*)alloc((size_t)S_ * S_ * 2);
    unsigned short* UhTlo = (unsigned short*)alloc((size_t)S_ * S_ * 2);
    unsigned short* Wohi  = (unsigned short*)alloc((size_t)E_ * S_ * 2);
    unsigned short* Wolo  = (unsigned short*)alloc((size_t)E_ * S_ * 2);
    unsigned short* zhi   = (unsigned short*)alloc((size_t)M_ * S_ * 2);
    unsigned short* zlo   = (unsigned short*)alloc((size_t)M_ * S_ * 2);
    float*          zu    = (float*)alloc((size_t)M_ * S_ * 4);
    unsigned short* shi   = (unsigned short*)alloc((size_t)M_ * S_ * 2);
    unsigned short* slo   = (unsigned short*)alloc((size_t)M_ * S_ * 2);
    float*          h     = (float*)alloc((size_t)B_ * S_ * 4);
    unsigned short* hhi   = (unsigned short*)alloc((size_t)B_ * S_ * 2);
    unsigned short* hlo   = (unsigned short*)alloc((size_t)B_ * S_ * 2);
    float*          psum  = (float*)alloc(16 * NWG * 2 * 4);
    unsigned*       cnt   = (unsigned*)alloc(256);

    // ---- precision prep: fp32 -> bf16 hi/lo splits (transpose U_z, U_h)
    split_kernel<<<2048, 256, 0, stream>>>(x, Xhi, Xlo, (size_t)M_ * E_);
    split_kernel<<<1024, 256, 0, stream>>>(Wz, Wzhi, Wzlo, (size_t)S_ * E_);
    split_kernel<<<1024, 256, 0, stream>>>(Wout, Wohi, Wolo, (size_t)E_ * S_);
    split_transpose_kernel<<<4096, 256, 0, stream>>>(U_z, UzThi, UzTlo, S_, S_);
    split_transpose_kernel<<<4096, 256, 0, stream>>>(U_h, UhThi, UhTlo, S_, S_);
    init_h_kernel<<<64, 256, 0, stream>>>(h, hhi, hlo, cnt);

    const int gemmGrid = (M_ / 64) * (S_ / 128);   // 4096 CTAs

    // ---- z = tanh(X @ Wz^T + bz)  -> bf16 hi/lo
    gemm_bf16x3<0><<<gemmGrid, 256, 0, stream>>>(Xhi, Xlo, Wzhi, Wzlo, bz,
                                                 nullptr, zhi, zlo, S_, E_);
    // ---- zu = z @ U_z + b_u  -> fp32
    gemm_bf16x3<1><<<gemmGrid, 256, 0, stream>>>(zhi, zlo, UzThi, UzTlo, b_u,
                                                 zu, nullptr, nullptr, S_, S_);
    // ---- sequential scan: persistent kernel, U_h^T slice cached in LDS
    const size_t recSmem = (size_t)2 * 64 * BSTRIDE * sizeof(unsigned short);
    recurrence_kernel<<<NWG, 128, recSmem, stream>>>(UhThi, UhTlo, zhi, zlo, zu,
                                                     g_s, b_s, h, hhi, hlo,
                                                     shi, slo, psum, cnt);
    // ---- out = states @ Wout^T + bout -> fp32 (pre-LN) in d_out
    gemm_bf16x3<1><<<gemmGrid, 256, 0, stream>>>(shi, slo, Wohi, Wolo, bout,
                                                 out, nullptr, nullptr, E_, S_);
    // ---- final layernorm over E, in place
    layernorm_rows<<<M_, 256, 0, stream>>>(out, g_o, b_o);
}
